// EnhancedRPTModel_77515569758930
// MI455X (gfx1250) — matrix operand order repaired and checked by reading, not compiled, not run on hardware
//
#include <hip/hip_runtime.h>
#include <hip/hip_bf16.h>
#include <stdint.h>

// ---------------------------------------------------------------------------
// Types for WMMA
// ---------------------------------------------------------------------------
typedef __attribute__((ext_vector_type(16))) _Float16 v16h;
typedef __attribute__((ext_vector_type(8)))  _Float16 v8h;
typedef __attribute__((ext_vector_type(4)))  _Float16 v4h;
typedef __attribute__((ext_vector_type(8)))  float    v8f;

#define TILE_M 128
#define TILE_N 128
#define TILE_K 32
#define LDS_LD 40   // row stride in halves (pad vs bank conflicts, keeps 16B align)

// ---------------------------------------------------------------------------
// Batched WMMA GEMM (dimension-aligned: requires M%128==0, N%128==0, K%32==0 —
// true for every matmul in this network except the tiny gate, handled apart):
//   C[z] = beta*C[z] + act( alpha * rowScale[m] * (A[z] @ B[z] + bias) )
// A: f32 [M,K] + lda/batch strides; B row-major [K,N] (bT=0) or [N,K] (bT=1).
// Batch z = (bb = z/batH, hh = z%batH) with separate strides -> addresses
// per-(batch,head) attention sub-matrices without explicit transposes.
// 8 wave32 waves per 256-thread block; each wave owns a 32x64 tile computed
// as 2x4 tiles of v_wmma_f32_16x16x32_f16.
// Register-double-buffered staging: tile t+1's 8x global_load_b128 are issued
// right after the barrier so their latency overlaps tile t's WMMA block.
// ---------------------------------------------------------------------------
__global__ __launch_bounds__(256) void wmma_gemm_kernel(
    const float* __restrict__ A, long long lda, long long sAb, long long sAh,
    const float* __restrict__ B, long long ldb, long long sBb, long long sBh, int bT,
    const float* __restrict__ bias,
    const float* __restrict__ rowScale, long long rsStride,
    float* __restrict__ C, long long ldc, long long sCb, long long sCh,
    int K, float alpha, int beta, int act, int batH)
{
    __shared__ _Float16 As[TILE_M][LDS_LD];
    __shared__ _Float16 Bs[TILE_N][LDS_LD];   // stored transposed: Bs[n][k]

    const int tid  = threadIdx.x;
    const int lane = tid & 31;
    const int wid  = tid >> 5;
    const int wm   = (wid & 3) * 32;   // wave M offset inside block tile
    const int wn   = (wid >> 2) * 64;  // wave N offset inside block tile
    const int m0   = blockIdx.y * TILE_M;
    const int n0   = blockIdx.x * TILE_N;
    const int z    = blockIdx.z;
    const int bb   = z / batH;
    const int hh   = z % batH;

    const float* Ap = A + (long long)bb * sAb + (long long)hh * sAh;
    const float* Bp = B + (long long)bb * sBb + (long long)hh * sBh;
    float*       Cp = C + (long long)bb * sCb + (long long)hh * sCh;

    v8f acc[2][4];
#pragma unroll
    for (int i = 0; i < 2; ++i)
#pragma unroll
        for (int j = 0; j < 4; ++j)
            acc[i][j] = (v8f){0.f,0.f,0.f,0.f,0.f,0.f,0.f,0.f};

    const int halfsel = lane >> 4;   // 0: lanes 0-15, 1: lanes 16-31
    const int sublane = lane & 15;

    // Per-thread staging coordinates (loop-invariant)
    const int aR  = tid >> 3;             // A row 0..127 (i adds 32)
    const int aC4 = (tid & 7) << 2;       // A col 0,4,..28
    const int bR0 = tid >> 5;             // bT=0: k row 0..7 (i adds 8)
    const int bC4 = (tid & 31) << 2;      // bT=0: n col 0..124
    const int bN  = tid >> 3;             // bT=1: n row 0..127 (i adds 32)
    const int bK4 = (tid & 7) << 2;       // bT=1: k col 0,4,..28

    union UA { v16h v; v8h h[2]; };

    float4 ra[4], rb[4];

    // Issue global loads for K-tile starting at k0 (8x global_load_b128).
    auto load_tile = [&](int k0) {
#pragma unroll
        for (int i = 0; i < 4; ++i)
            ra[i] = *reinterpret_cast<const float4*>(
                Ap + (long long)(m0 + aR + 32 * i) * lda + (k0 + aC4));
        if (bT == 0) {
#pragma unroll
            for (int i = 0; i < 4; ++i)
                rb[i] = *reinterpret_cast<const float4*>(
                    Bp + (long long)(k0 + bR0 + 8 * i) * ldb + (n0 + bC4));
        } else {
#pragma unroll
            for (int i = 0; i < 4; ++i)
                rb[i] = *reinterpret_cast<const float4*>(
                    Bp + (long long)(n0 + bN + 32 * i) * ldb + (k0 + bK4));
        }
    };

    // Convert staged registers to f16 and write into LDS.
    auto store_tile = [&]() {
#pragma unroll
        for (int i = 0; i < 4; ++i) {
            v4h h; h.x = (_Float16)ra[i].x; h.y = (_Float16)ra[i].y;
                   h.z = (_Float16)ra[i].z; h.w = (_Float16)ra[i].w;
            *reinterpret_cast<v4h*>(&As[aR + 32 * i][aC4]) = h;
        }
        if (bT == 0) {
#pragma unroll
            for (int i = 0; i < 4; ++i) {
                int r = bR0 + 8 * i;
                Bs[bC4 + 0][r] = (_Float16)rb[i].x;
                Bs[bC4 + 1][r] = (_Float16)rb[i].y;
                Bs[bC4 + 2][r] = (_Float16)rb[i].z;
                Bs[bC4 + 3][r] = (_Float16)rb[i].w;
            }
        } else {
#pragma unroll
            for (int i = 0; i < 4; ++i) {
                v4h h; h.x = (_Float16)rb[i].x; h.y = (_Float16)rb[i].y;
                       h.z = (_Float16)rb[i].z; h.w = (_Float16)rb[i].w;
                *reinterpret_cast<v4h*>(&Bs[bN + 32 * i][bK4]) = h;
            }
        }
    };

    const int tiles = K >> 5;        // K % 32 == 0
    load_tile(0);                    // preload tile 0

    for (int t = 0; t < tiles; ++t) {
        store_tile();                // regs -> LDS (waits on outstanding loads)
        __syncthreads();

        // Prefetch next tile's global data into registers: these loads are in
        // flight while the WMMA block below executes.
        if (t + 1 < tiles) load_tile((t + 1) * TILE_K);

        // ---- load fragments & issue WMMAs (K=32 per staging) ----
        UA af[2];
#pragma unroll
        for (int mi = 0; mi < 2; ++mi) {
            int row = wm + mi * 16 + sublane;
            int kb = halfsel * 8;                     // A ISA layout: K groups of 8
            af[mi].h[0] = *(const v8h*)&As[row][kb];
            af[mi].h[1] = *(const v8h*)&As[row][16 + kb];
        }
#pragma unroll
        for (int ni = 0; ni < 4; ++ni) {
            int col = wn + ni * 16 + sublane;
            int ks = halfsel * 8;                     // B ISA layout: K groups of 8
            UA bf;
            bf.h[0] = *(const v8h*)&Bs[col][ks];
            bf.h[1] = *(const v8h*)&Bs[col][16 + ks];
#pragma unroll
            for (int mi = 0; mi < 2; ++mi) {
                acc[mi][ni] = __builtin_amdgcn_wmma_f32_16x16x32_f16(
                    false, af[mi].v, false, bf.v, (short)0, acc[mi][ni], false, false);
            }
        }
        __syncthreads();
    }

    // ---- epilogue: bias, rowScale, alpha, act, beta accumulate ----
#pragma unroll
    for (int mi = 0; mi < 2; ++mi) {
#pragma unroll
        for (int ni = 0; ni < 4; ++ni) {
            int nIdx = n0 + wn + ni * 16 + sublane;
            float bv = bias ? bias[nIdx] : 0.f;
            int mBase = m0 + wm + mi * 16 + halfsel * 8;
#pragma unroll
            for (int j = 0; j < 8; ++j) {
                int m = mBase + j;
                float tv = acc[mi][ni][j] + bv;
                if (rowScale) tv *= rowScale[(long long)m * rsStride];
                tv *= alpha;
                if (act == 1) tv = fmaxf(tv, 0.f);
                float* cp = Cp + (long long)m * ldc + nIdx;
                if (beta) tv += *cp;
                *cp = tv;
            }
        }
    }
}

// ---------------------------------------------------------------------------
// Fused MoE gate: logits = x@gate_w + gate_b (E=8), softmax, top-2, re-softmax,
// dense combine weights. One 256-thread block per token; 8 running dots per
// thread over K=H, shared-memory reduce, lane 0 finishes routing.
// ---------------------------------------------------------------------------
__global__ __launch_bounds__(256) void gate_fused_kernel(
    const float* __restrict__ X, const float* __restrict__ gw,
    const float* __restrict__ gb, float* __restrict__ comb, int H)
{
    int n = blockIdx.x;
    int tid = threadIdx.x;
    const float* xr = X + (long long)n * H;

    float acc[8] = {0.f,0.f,0.f,0.f,0.f,0.f,0.f,0.f};
    for (int c = tid; c < H; c += 256) {
        float x = xr[c];
        const float* wr = gw + (long long)c * 8;
#pragma unroll
        for (int e = 0; e < 8; ++e) acc[e] += x * wr[e];
    }

    __shared__ float red[256];
    __shared__ float logit[8];
    for (int e = 0; e < 8; ++e) {
        red[tid] = acc[e]; __syncthreads();
        for (int k = 128; k > 0; k >>= 1) { if (tid < k) red[tid] += red[tid + k]; __syncthreads(); }
        if (tid == 0) logit[e] = red[0] + gb[e];
        __syncthreads();
    }

    if (tid == 0) {
        float l[8];
        float mx = -3.4e38f;
#pragma unroll
        for (int e = 0; e < 8; ++e) { l[e] = logit[e]; mx = fmaxf(mx, l[e]); }
        float s = 0.f;
#pragma unroll
        for (int e = 0; e < 8; ++e) { l[e] = __expf(l[e] - mx); s += l[e]; }
#pragma unroll
        for (int e = 0; e < 8; ++e) l[e] /= s;           // gate_probs
        int i1 = 0;
#pragma unroll
        for (int e = 1; e < 8; ++e) if (l[e] > l[i1]) i1 = e;
        int i2 = (i1 == 0) ? 1 : 0;
#pragma unroll
        for (int e = 0; e < 8; ++e) if (e != i1 && l[e] > l[i2]) i2 = e;
        float m2 = fmaxf(l[i1], l[i2]);
        float e1 = __expf(l[i1] - m2), e2 = __expf(l[i2] - m2);
        float inv = 1.f / (e1 + e2);
#pragma unroll
        for (int e = 0; e < 8; ++e) comb[n * 8 + e] = 0.f;
        comb[n * 8 + i1] = e1 * inv;
        comb[n * 8 + i2] = e2 * inv;
    }
}

// ---------------------------------------------------------------------------
// Row softmax (in place). Optional mask term: v += mask[row/rowsPerBatch][c]*-1e9
// ---------------------------------------------------------------------------
__global__ __launch_bounds__(256) void softmax_rows_kernel(
    float* __restrict__ data, int rows, int cols,
    const float* __restrict__ mask, int rowsPerBatch)
{
    int row = blockIdx.x;
    if (row >= rows) return;
    float* p = data + (long long)row * cols;
    const float* mrow = mask ? (mask + (long long)(row / rowsPerBatch) * cols) : nullptr;
    __shared__ float red[256];
    int tid = threadIdx.x;

    float mx = -3.4e38f;
    for (int c = tid; c < cols; c += 256) {
        float v = p[c];
        if (mrow) v += mrow[c] * (-1e9f);
        mx = fmaxf(mx, v);
    }
    red[tid] = mx; __syncthreads();
    for (int s = 128; s > 0; s >>= 1) { if (tid < s) red[tid] = fmaxf(red[tid], red[tid + s]); __syncthreads(); }
    mx = red[0]; __syncthreads();

    float sum = 0.f;
    for (int c = tid; c < cols; c += 256) {
        float v = p[c];
        if (mrow) v += mrow[c] * (-1e9f);
        float e = __expf(v - mx);
        p[c] = e;
        sum += e;
    }
    red[tid] = sum; __syncthreads();
    for (int s = 128; s > 0; s >>= 1) { if (tid < s) red[tid] += red[tid + s]; __syncthreads(); }
    float inv = 1.f / red[0];
    for (int c = tid; c < cols; c += 256) p[c] *= inv;
}

// ---------------------------------------------------------------------------
// LayerNorm rows (in place), per-row mean/var, affine g/b.
// ---------------------------------------------------------------------------
__global__ __launch_bounds__(256) void layernorm_kernel(
    float* __restrict__ data, const float* __restrict__ g, const float* __restrict__ b,
    int rows, int cols)
{
    int row = blockIdx.x;
    if (row >= rows) return;
    float* p = data + (long long)row * cols;
    __shared__ float red[256];
    int tid = threadIdx.x;

    float s = 0.f;
    for (int c = tid; c < cols; c += 256) s += p[c];
    red[tid] = s; __syncthreads();
    for (int k = 128; k > 0; k >>= 1) { if (tid < k) red[tid] += red[tid + k]; __syncthreads(); }
    float mean = red[0] / (float)cols; __syncthreads();

    float v = 0.f;
    for (int c = tid; c < cols; c += 256) { float d = p[c] - mean; v += d * d; }
    red[tid] = v; __syncthreads();
    for (int k = 128; k > 0; k >>= 1) { if (tid < k) red[tid] += red[tid + k]; __syncthreads(); }
    float inv = rsqrtf(red[0] / (float)cols + 1e-5f);

    for (int c = tid; c < cols; c += 256) p[c] = (p[c] - mean) * inv * g[c] + b[c];
}

// ---------------------------------------------------------------------------
// Per-row dot with small vector + sigmoid: out[row] = sigmoid(dot(X[row],w)+b)
// ---------------------------------------------------------------------------
__global__ __launch_bounds__(256) void rowdot_sigmoid_kernel(
    const float* __restrict__ X, const float* __restrict__ w,
    const float* __restrict__ bscal, float* __restrict__ out, int cols)
{
    int row = blockIdx.x;
    __shared__ float red[256];
    int tid = threadIdx.x;
    float s = 0.f;
    for (int c = tid; c < cols; c += 256) s += X[(long long)row * cols + c] * w[c];
    red[tid] = s; __syncthreads();
    for (int k = 128; k > 0; k >>= 1) { if (tid < k) red[tid] += red[tid + k]; __syncthreads(); }
    if (tid == 0) out[row] = 1.f / (1.f + __expf(-(red[0] + bscal[0])));
}

// ---------------------------------------------------------------------------
// cur += g[n]*so ; also write cur into concat[:, stepOff:stepOff+H]
// ---------------------------------------------------------------------------
__global__ void gated_update_kernel(float* __restrict__ cur, const float* __restrict__ so,
                                    const float* __restrict__ gv, float* __restrict__ concat,
                                    int Ntok, int H, int stepOff, int concatLd)
{
    long long idx = (long long)blockIdx.x * blockDim.x + threadIdx.x;
    if (idx >= (long long)Ntok * H) return;
    int n = (int)(idx / H);
    int c = (int)(idx % H);
    float v = cur[idx] + gv[n] * so[idx];
    cur[idx] = v;
    concat[(long long)n * concatLd + stepOff + c] = v;
}

// a[i] += s * b[i]
__global__ void axpy_kernel(float* __restrict__ a, const float* __restrict__ b,
                            float s, long long n)
{
    long long idx = (long long)blockIdx.x * blockDim.x + threadIdx.x;
    if (idx < n) a[idx] += s * b[idx];
}

// ---------------------------------------------------------------------------
// Host-side launch helper
// ---------------------------------------------------------------------------
static inline void launch_gemm(hipStream_t st,
    const float* A, long long lda, long long sAb, long long sAh,
    const float* B, long long ldb, long long sBb, long long sBh, int bT,
    const float* bias, const float* rowScale, long long rsStride,
    float* C, long long ldc, long long sCb, long long sCh,
    int M, int Ncols, int K, float alpha, int beta, int act,
    int batches, int batH)
{
    dim3 grid(Ncols / TILE_N, M / TILE_M, batches);
    wmma_gemm_kernel<<<grid, dim3(256, 1, 1), 0, st>>>(
        A, lda, sAb, sAh, B, ldb, sBb, sBh, bT,
        bias, rowScale, rsStride, C, ldc, sCb, sCh,
        K, alpha, beta, act, batH);
}

extern "C" void kernel_launch(void* const* d_in, const int* in_sizes, int n_in,
                              void* d_out, int out_size, void* d_ws, size_t ws_size,
                              hipStream_t stream) {
    (void)in_sizes; (void)n_in; (void)out_size; (void)ws_size;

    // Problem constants
    const int Bb = 2, S = 2048, H = 2048, E = 8, HID = 4096;
    const int NH = 8, HD = 256, MS = 256, MD = 512, RD = 512;
    const int N = Bb * S;                     // 4096 tokens
    const long long LL = 1;

    // Inputs (setup_inputs order)
    const float* hidden     = (const float*)d_in[0];
    const float* attn_mask  = (const float*)d_in[1];
    const float* gate_w     = (const float*)d_in[2];
    const float* gate_b     = (const float*)d_in[3];
    const float* moe_w1     = (const float*)d_in[4];
    const float* moe_b1     = (const float*)d_in[5];
    const float* moe_w2     = (const float*)d_in[6];
    const float* moe_b2     = (const float*)d_in[7];
    const float* q_w        = (const float*)d_in[8];
    const float* q_b        = (const float*)d_in[9];
    const float* k_w        = (const float*)d_in[10];
    const float* k_b        = (const float*)d_in[11];
    const float* v_w        = (const float*)d_in[12];
    const float* v_b        = (const float*)d_in[13];
    const float* o_w        = (const float*)d_in[14];
    const float* o_b        = (const float*)d_in[15];
    const float* mem_values = (const float*)d_in[16];
    const float* mem_proj_w = (const float*)d_in[17];
    const float* mem_proj_b = (const float*)d_in[18];
    const float* mem_attn_w = (const float*)d_in[19];
    const float* mem_attn_b = (const float*)d_in[20];
    const float* rs_w1      = (const float*)d_in[21];
    const float* rs_b1      = (const float*)d_in[22];
    const float* rs_w2      = (const float*)d_in[23];
    const float* rs_b2      = (const float*)d_in[24];
    const float* ln_g       = (const float*)d_in[25];
    const float* ln_b       = (const float*)d_in[26];
    const float* hg_w1      = (const float*)d_in[27];
    const float* hg_b1      = (const float*)d_in[28];
    const float* hg_w2      = (const float*)d_in[29];
    const float* hg_b2      = (const float*)d_in[30];
    const float* integ_w    = (const float*)d_in[31];
    const float* integ_b    = (const float*)d_in[32];
    float* out = (float*)d_out;

    // Workspace layout (float offsets, MF = 1M floats)
    const long long MF = 1ll << 20;
    float* ws     = (float*)d_ws;
    float* hbuf   = ws;                        // [N,H]        8M
    float* comb   = ws + 8 * MF;               // [N,E] combine weights
    float* h1     = ws + 9 * MF;               // [N,HID]      16M (dead after MoE)
    float* qb     = ws + 9 * MF;               // alias h1 lo  8M
    float* kb     = ws + 17 * MF;              // alias h1 hi  8M
    float* vb     = ws + 25 * MF;              // 8M
    float* scores = ws + 33 * MF;              // [B,NH,S,S]   64M
    float* attn   = ws + 97 * MF;              // [N,H]        8M
    float* memlog = ws + 105 * MF;             // [N,MS]       1M
    float* memmid = ws + 106 * MF;             // [N,MD]       2M
    float* memo   = ws + 108 * MF;             // [N,H]        8M
    float* cur    = ws + 116 * MF;             // [N,H]        8M
    float* sob    = ws + 124 * MF;             // [N,H]        8M
    float* t1     = ws + 132 * MF;             // [N,RD]       2M
    float* g1     = ws + 134 * MF;             // [N,RD]       2M
    float* gv     = ws + 136 * MF;             // [N]
    float* concat = scores;                    // [N,3H] 24M, aliases dead scores

    const long long NH_ll = (long long)N * H;

    // ---- residual base: h = hidden_states ----
    hipMemcpyAsync(hbuf, hidden, (size_t)NH_ll * sizeof(float),
                   hipMemcpyDeviceToDevice, stream);

    // ---- MoE gate (fused logits + softmax + top-2 + combine weights) ----
    gate_fused_kernel<<<N, 256, 0, stream>>>(hidden, gate_w, gate_b, comb, H);

    // ---- MoE experts: h += 0.5 * comb[:,e] * (relu(x@W1e+b1e) @ W2e + b2e) ----
    for (int e = 0; e < E; ++e) {
        launch_gemm(stream, hidden, H, 0, 0,
                    moe_w1 + LL * e * H * HID, HID, 0, 0, 0,
                    moe_b1 + LL * e * HID, nullptr, 0,
                    h1, HID, 0, 0, N, HID, H, 1.f, 0, 1, 1, 1);
        launch_gemm(stream, h1, HID, 0, 0,
                    moe_w2 + LL * e * HID * H, H, 0, 0, 0,
                    moe_b2 + LL * e * H, comb + e, E,
                    hbuf, H, 0, 0, N, H, HID, 0.5f, 1, 0, 1, 1);
    }

    // ---- attention projections ----
    launch_gemm(stream, hbuf, H, 0, 0, q_w, H, 0, 0, 0, q_b, nullptr, 0,
                qb, H, 0, 0, N, H, H, 1.f, 0, 0, 1, 1);
    launch_gemm(stream, hbuf, H, 0, 0, k_w, H, 0, 0, 0, k_b, nullptr, 0,
                kb, H, 0, 0, N, H, H, 1.f, 0, 0, 1, 1);
    launch_gemm(stream, hbuf, H, 0, 0, v_w, H, 0, 0, 0, v_b, nullptr, 0,
                vb, H, 0, 0, N, H, H, 1.f, 0, 0, 1, 1);

    // ---- scores[b,h] = Q_bh @ K_bh^T / 16 ----
    launch_gemm(stream,
                qb, H, (long long)S * H, HD,
                kb, H, (long long)S * H, HD, /*bT=*/1,
                nullptr, nullptr, 0,
                scores, S, (long long)NH * S * S, (long long)S * S,
                S, S, HD, 1.f / 16.f, 0, 0, Bb * NH, NH);

    // ---- softmax over keys with mask term ----
    softmax_rows_kernel<<<Bb * NH * S, 256, 0, stream>>>(
        scores, Bb * NH * S, S, attn_mask, NH * S);

    // ---- attn[b,h] = probs @ V_bh, written back to [B,S,NH*HD] layout ----
    launch_gemm(stream,
                scores, S, (long long)NH * S * S, (long long)S * S,
                vb, H, (long long)S * H, HD, /*bT=*/0,
                nullptr, nullptr, 0,
                attn, H, (long long)S * H, HD,
                S, HD, S, 1.f, 0, 0, Bb * NH, NH);

    // ---- memory branch ----
    launch_gemm(stream, hbuf, H, 0, 0, mem_attn_w, MS, 0, 0, 0,
                mem_attn_b, nullptr, 0, memlog, MS, 0, 0,
                N, MS, H, 1.f, 0, 0, 1, 1);
    softmax_rows_kernel<<<N, 256, 0, stream>>>(memlog, N, MS, nullptr, 1);
    launch_gemm(stream, memlog, MS, 0, 0, mem_values, MD, 0, 0, 0,
                nullptr, nullptr, 0, memmid, MD, 0, 0,
                N, MD, MS, 1.f, 0, 0, 1, 1);
    launch_gemm(stream, memmid, MD, 0, 0, mem_proj_w, H, 0, 0, 0,
                mem_proj_b, nullptr, 0, memo, H, 0, 0,
                N, H, MD, 1.f, 0, 0, 1, 1);

    // ---- combined = attn + 0.3*mem_o ; h += combined @ o_w + o_b ----
    axpy_kernel<<<(int)((NH_ll + 255) / 256), 256, 0, stream>>>(attn, memo, 0.3f, NH_ll);
    launch_gemm(stream, attn, H, 0, 0, o_w, H, 0, 0, 0, o_b, nullptr, 0,
                hbuf, H, 0, 0, N, H, H, 1.f, 1, 0, 1, 1);

    // ---- hierarchical reasoning ----
    hipMemcpyAsync(cur, hbuf, (size_t)NH_ll * sizeof(float),
                   hipMemcpyDeviceToDevice, stream);
    for (int i = 0; i < 3; ++i) {
        launch_gemm(stream, cur, H, 0, 0,
                    rs_w1 + LL * i * H * RD, RD, 0, 0, 0,
                    rs_b1 + LL * i * RD, nullptr, 0,
                    t1, RD, 0, 0, N, RD, H, 1.f, 0, 1, 1, 1);
        launch_gemm(stream, t1, RD, 0, 0,
                    rs_w2 + LL * i * RD * H, H, 0, 0, 0,
                    rs_b2 + LL * i * H, nullptr, 0,
                    sob, H, 0, 0, N, H, RD, 1.f, 0, 0, 1, 1);
        layernorm_kernel<<<N, 256, 0, stream>>>(sob, ln_g + LL * i * H, ln_b + LL * i * H, N, H);
        launch_gemm(stream, cur, H, 0, 0,
                    hg_w1 + LL * i * H * (H / 4), H / 4, 0, 0, 0,
                    hg_b1 + LL * i * (H / 4), nullptr, 0,
                    g1, H / 4, 0, 0, N, H / 4, H, 1.f, 0, 1, 1, 1);
        rowdot_sigmoid_kernel<<<N, 256, 0, stream>>>(g1, hg_w2 + LL * i * (H / 4),
                                                     hg_b2 + i, gv, H / 4);
        gated_update_kernel<<<(int)((NH_ll + 255) / 256), 256, 0, stream>>>(
            cur, sob, gv, concat, N, H, i * H, 3 * H);
    }

    // ---- out = h + concat @ integ_w + integ_b ----
    hipMemcpyAsync(out, hbuf, (size_t)NH_ll * sizeof(float),
                   hipMemcpyDeviceToDevice, stream);
    launch_gemm(stream, concat, 3 * H, 0, 0, integ_w, H, 0, 0, 0,
                integ_b, nullptr, 0, out, H, 0, 0,
                N, H, 3 * H, 1.f, 1, 0, 1, 1);
}
